// CommNetActor_403726926511
// MI455X (gfx1250) — matrix operand order repaired and checked by reading, not compiled, not run on hardware
//
#include <hip/hip_runtime.h>
#include <hip/hip_bf16.h>

// CommNet actor on gfx1250: all GEMMs via v_wmma_f32_16x16x32_f16 (wave32).
// Dominant op adj@h is HBM-bound (256MB adj @ 23.3TB/s ~= 11.5us); f16 WMMA
// with f32 accumulation keeps it there. adj in {0,1} casts to f16 exactly.
// The message kernel stages the hT tile with GLOBAL_LOAD_ASYNC_TO_LDS_B128
// (ASYNCcnt-tracked) and software-pipelines the adj int->f16 conversion two
// K-steps deep across a 3-buffer LDS rotation.

#define N_AGENTS 8192
#define OBS_DIM  64
#define HID      128
#define ACT_DIM  16

typedef _Float16 v16h __attribute__((ext_vector_type(16)));
typedef _Float16 v8h  __attribute__((ext_vector_type(8)));
typedef float    v8f  __attribute__((ext_vector_type(8)));

static __device__ __forceinline__ v16h cat8(v8h lo, v8h hi) {
  return __builtin_shufflevector(lo, hi, 0,1,2,3,4,5,6,7,8,9,10,11,12,13,14,15);
}

static __device__ __forceinline__ v8f wmma32(v16h a, v16h b, v8f c) {
  // (neg_a, A, neg_b, B, c_mod, C, reuse_a, reuse_b)
  return __builtin_amdgcn_wmma_f32_16x16x32_f16(false, a, false, b, (short)0, c,
                                                false, false);
}

static __device__ __forceinline__ v8f vzero8() {
  v8f z = {0.f, 0.f, 0.f, 0.f, 0.f, 0.f, 0.f, 0.f};
  return z;
}

// A fragment (16x32 f16) from a row-major f16 buffer.
// p = &src[row*ld + k0 + (lane>>4)*8]; elements 0..7 <- K kof..kof+7,
// elements 8..15 <- K kof+16..kof+23 (per ISA 16-bit A layout).
static __device__ __forceinline__ v16h afrag_f16(const _Float16* p) {
  v8h lo = *(const v8h*)p;
  v8h hi = *(const v8h*)(p + 16);
  return cat8(lo, hi);
}

// A fragment from row-major f32 buffer with on-the-fly f16 convert.
static __device__ __forceinline__ v16h afrag_f32(const float* p) {
  v16h r;
#pragma unroll
  for (int i = 0; i < 8; ++i) r[i] = (_Float16)p[i];
#pragma unroll
  for (int i = 0; i < 8; ++i) r[8 + i] = (_Float16)p[16 + i];
  return r;
}

// B fragment (32x16 f16) from a K-major (transposed) buffer.
// p = &wT[n*ldk + k0 + (lane>>4)*8]; lane gives column N; elements 0..7 <- K kof..kof+7, 8..15 <- K kof+16..kof+23.
static __device__ __forceinline__ v16h bfrag_kmaj(const _Float16* p) {
  v8h lo = *(const v8h*)p;
  v8h hi = *(const v8h*)(p + 16);
  return cat8(lo, hi);
}

// Async global->LDS copy of 16 bytes (per lane), tracked on ASYNCcnt.
static __device__ __forceinline__ void async_copy_b128(unsigned lds_byte_off,
                                                       const void* gaddr) {
  asm volatile("global_load_async_to_lds_b128 %0, %1, off"
               :
               : "v"(lds_byte_off), "v"(gaddr)
               : "memory");
}
// Allow the newest 2 async ops (this iteration's fill) to remain in flight;
// async loads complete in order, so older buffers are guaranteed landed.
static __device__ __forceinline__ void wait_async_le2() {
  asm volatile("s_wait_asynccnt 0x2" ::: "memory");
}

// ---------------------------------------------------------------------------
// Kernel 0: transpose+convert weights to f16 K-major (B-fragment friendly).
// ---------------------------------------------------------------------------
__global__ __launch_bounds__(256) void k_prep(const float* __restrict__ W1,
                                              const float* __restrict__ W2,
                                              const float* __restrict__ W3,
                                              _Float16* __restrict__ w1t,
                                              _Float16* __restrict__ w2t,
                                              _Float16* __restrict__ w3t) {
  const int stride = gridDim.x * blockDim.x;
  const int i0 = blockIdx.x * blockDim.x + threadIdx.x;
  for (int i = i0; i < HID * OBS_DIM; i += stride) {       // w1t[f][k]
    int f = i / OBS_DIM, k = i % OBS_DIM;
    w1t[i] = (_Float16)W1[k * HID + f];
  }
  for (int i = i0; i < HID * 2 * HID; i += stride) {       // w2t[f][k]
    int f = i / (2 * HID), k = i % (2 * HID);
    w2t[i] = (_Float16)W2[k * HID + f];
  }
  for (int i = i0; i < ACT_DIM * HID; i += stride) {       // w3t[a][k]
    int a = i / HID, k = i % HID;
    w3t[i] = (_Float16)W3[k * ACT_DIM + a];
  }
}

// ---------------------------------------------------------------------------
// Kernel 1: h = tanh(obs @ W1 + b1). Writes h row-major f16 and hT [HID][N]
// f16 (LDS transpose epilogue, so message kernel B loads are contiguous).
// Grid: 128 blocks x 64 rows, 8 waves; wave -> (M-tile = w>>1, N-half = w&1).
// ---------------------------------------------------------------------------
__global__ __launch_bounds__(256) void k_encoder(const float* __restrict__ obs,
                                                 const _Float16* __restrict__ w1t,
                                                 const float* __restrict__ b1,
                                                 _Float16* __restrict__ hrow,
                                                 _Float16* __restrict__ hT) {
  __shared__ _Float16 tile[64][HID + 2];  // +2 pad: agents stride 130 halves
  const int tid = threadIdx.x, lane = tid & 31, wave = tid >> 5;
  const int mtile = wave >> 1, ng = (wave & 1) * 64;
  const int hi = lane >> 4, r16 = lane & 15;
  const int mBlock = blockIdx.x * 64;
  const int row = mBlock + mtile * 16 + r16;

  v8f acc[4];
#pragma unroll
  for (int j = 0; j < 4; ++j) acc[j] = vzero8();

#pragma unroll
  for (int s = 0; s < OBS_DIM / 32; ++s) {
    const int k0 = s * 32;
    v16h a = afrag_f32(obs + (size_t)row * OBS_DIM + k0 + hi * 8);
#pragma unroll
    for (int j = 0; j < 4; ++j) {
      const int n = ng + j * 16 + r16;
      v16h b = bfrag_kmaj(w1t + n * OBS_DIM + k0 + hi * 8);
      acc[j] = wmma32(a, b, acc[j]);
    }
  }

  // epilogue: bias + tanh into LDS tile [agent][feat]
#pragma unroll
  for (int j = 0; j < 4; ++j) {
    const int col = ng + j * 16 + r16;
    const float bv = b1[col];
#pragma unroll
    for (int rr = 0; rr < 8; ++rr) {
      tile[mtile * 16 + hi * 8 + rr][col] = (_Float16)tanhf(acc[j][rr] + bv);
    }
  }
  __syncthreads();

  {  // coalesced row-major h store: thread -> (agent, 32-col chunk)
    const int lr = tid >> 2, c0 = (tid & 3) * 32;
    _Float16* dst = hrow + (size_t)(mBlock + lr) * HID + c0;
    const _Float16* src = &tile[lr][c0];
#pragma unroll
    for (int i = 0; i < 32; ++i) dst[i] = src[i];
  }
  {  // coalesced hT store: thread -> (feat, 32-agent chunk)
    const int f = tid >> 1, a0 = (tid & 1) * 32;
    _Float16* dst = hT + (size_t)f * N_AGENTS + mBlock + a0;
#pragma unroll
    for (int i = 0; i < 32; ++i) dst[i] = tile[a0 + i][f];  // stride 130: conflict-free
  }
}

// ---------------------------------------------------------------------------
// Kernel 2 (dominant): msg = (adj @ h) / max(deg,1).
// C[8192x128] = A[8192x8192 int32 {0,1} -> f16] x B (hT-fed f16).
// 128 WGs x 64 output rows; K=8192 in steps of 32; 3-buffer LDS rotation:
//  - adj tile: global->reg at step s+3, reg->LDS (int->f16 + degree sum) at
//    s+2, computed at s  => ~1 iteration of load latency tolerance.
//  - hT tile: GLOBAL_LOAD_ASYNC_TO_LDS_B128 issued 2 steps ahead, gated by
//    s_wait_asynccnt<=2 before each barrier (in-order completion).
// ---------------------------------------------------------------------------
#define LDA 40  // 64x32 tile, padded stride (halves)
#define LDB 40  // 128x32 tile, padded stride (halves)
#define NSTEP (N_AGENTS / 32)

__global__ __launch_bounds__(256) void k_message(const int* __restrict__ adj,
                                                 const _Float16* __restrict__ hT,
                                                 _Float16* __restrict__ msgf) {
  __shared__ __align__(16) _Float16 sA[3][64 * LDA];
  __shared__ __align__(16) _Float16 sB[3][128 * LDB];
  __shared__ float sDeg[64];

  const int tid = threadIdx.x, lane = tid & 31, wave = tid >> 5;
  const int mtile = wave >> 1, ng = (wave & 1) * 64;
  const int hi = lane >> 4, r16 = lane & 15;
  const size_t mBlock = (size_t)blockIdx.x * 64;

  if (tid < 64) sDeg[tid] = 0.f;

  // staging assignment: 4 threads per adj row (8 ints each), 2 per hT row
  const int arow = tid >> 2, akc = (tid & 3) * 8;
  const int* aptr = adj + (mBlock + arow) * (size_t)N_AGENTS + akc;
  const int brow = tid >> 1, bkc = (tid & 1) * 16;
  const _Float16* bptr = hT + (size_t)brow * N_AGENTS + bkc;

  int degAcc = 0;
  int r[8];  // in-flight adj registers (one K-step ahead of its LDS store)
  v8f acc[4];
#pragma unroll
  for (int j = 0; j < 4; ++j) acc[j] = vzero8();

  // fill helpers (kept inline for clarity)
#define ISSUE_B(step, bi)                                                      \
  do {                                                                         \
    const _Float16* g_ = bptr + (size_t)(step) * 32;                           \
    unsigned l_ = (unsigned)(uintptr_t)&sB[(bi)][brow * LDB + bkc];            \
    async_copy_b128(l_, (const void*)g_);                                      \
    async_copy_b128(l_ + 16, (const void*)(g_ + 8));                           \
  } while (0)

#define LOAD_ADJ(step)                                                         \
  do {                                                                         \
    const int* p_ = aptr + (size_t)(step) * 32;                                \
    _Pragma("unroll") for (int i_ = 0; i_ < 8; ++i_) r[i_] = p_[i_];           \
  } while (0)

#define STORE_ADJ(bi)                                                          \
  do {                                                                         \
    _Pragma("unroll") for (int i_ = 0; i_ < 8; ++i_) {                         \
      degAcc += r[i_];                                                         \
      sA[(bi)][arow * LDA + akc + i_] = (_Float16)r[i_];                       \
    }                                                                          \
  } while (0)

  // prologue: fill buffers 0 and 1, preload regs for step 2
  ISSUE_B(0, 0);
  LOAD_ADJ(0);
  STORE_ADJ(0);
  ISSUE_B(1, 1);
  LOAD_ADJ(1);
  STORE_ADJ(1);
  LOAD_ADJ(2);
  wait_async_le2();  // buffer 0's async copies complete
  __syncthreads();

  for (int s = 0; s < NSTEP; ++s) {
    const int bi = s % 3;
    if (s + 2 < NSTEP) {
      const int fb = (s + 2) % 3;
      ISSUE_B(s + 2, fb);
      STORE_ADJ(fb);  // regs hold step s+2 (loaded last iteration)
      if (s + 3 < NSTEP) LOAD_ADJ(s + 3);
      if (s + 6 < NSTEP) __builtin_prefetch(aptr + (size_t)(s + 6) * 32, 0, 0);
    }
    // compute from buffer bi (published by previous iteration's barrier)
    const _Float16* ap = &sA[bi][(mtile * 16 + r16) * LDA + hi * 8];
    v16h a = cat8(*(const v8h*)ap, *(const v8h*)(ap + 16));
#pragma unroll
    for (int j = 0; j < 4; ++j) {
      const _Float16* bp = &sB[bi][(ng + j * 16 + r16) * LDB + hi * 8];
      acc[j] = wmma32(a, cat8(*(const v8h*)bp, *(const v8h*)(bp + 16)), acc[j]);
    }
    wait_async_le2();  // buffer (s+1)%3's async copies (issued at s-1) done
    __syncthreads();
  }

#undef ISSUE_B
#undef LOAD_ADJ
#undef STORE_ADJ

  atomicAdd(&sDeg[arow], (float)degAcc);
  __syncthreads();

#pragma unroll
  for (int rr = 0; rr < 8; ++rr) {
    const int lr = mtile * 16 + hi * 8 + rr;
    float d = sDeg[lr];
    if (d < 1.f) d = 1.f;
#pragma unroll
    for (int j = 0; j < 4; ++j) {
      const int col = ng + j * 16 + r16;
      msgf[(mBlock + lr) * HID + col] = (_Float16)(acc[j][rr] / d);
    }
  }
}

// ---------------------------------------------------------------------------
// Kernel 3: hid = tanh([h, msg] @ W2 + b2). Concat handled by picking the A
// source per K-tile; A/B fragments read directly from global (W2t is 64KB,
// L2/WGP$-resident). No barriers needed.
// ---------------------------------------------------------------------------
__global__ __launch_bounds__(256) void k_actor1(const _Float16* __restrict__ hrow,
                                                const _Float16* __restrict__ msgf,
                                                const _Float16* __restrict__ w2t,
                                                const float* __restrict__ b2,
                                                _Float16* __restrict__ hidf) {
  const int tid = threadIdx.x, lane = tid & 31, wave = tid >> 5;
  const int mtile = wave >> 1, ng = (wave & 1) * 64;
  const int hi = lane >> 4, r16 = lane & 15;
  const int mBlock = blockIdx.x * 64;
  const int row = mBlock + mtile * 16 + r16;

  v8f acc[4];
#pragma unroll
  for (int j = 0; j < 4; ++j) acc[j] = vzero8();

#pragma unroll
  for (int s = 0; s < (2 * HID) / 32; ++s) {
    const int k0 = s * 32;
    const _Float16* ap = (k0 < HID) ? (hrow + (size_t)row * HID + k0)
                                    : (msgf + (size_t)row * HID + (k0 - HID));
    v16h a = afrag_f16(ap + hi * 8);
#pragma unroll
    for (int j = 0; j < 4; ++j) {
      const int n = ng + j * 16 + r16;
      v16h b = bfrag_kmaj(w2t + n * (2 * HID) + k0 + hi * 8);
      acc[j] = wmma32(a, b, acc[j]);
    }
  }

#pragma unroll
  for (int j = 0; j < 4; ++j) {
    const int col = ng + j * 16 + r16;
    const float bv = b2[col];
#pragma unroll
    for (int rr = 0; rr < 8; ++rr) {
      const int gr = mBlock + mtile * 16 + hi * 8 + rr;
      hidf[(size_t)gr * HID + col] = (_Float16)tanhf(acc[j][rr] + bv);
    }
  }
}

// ---------------------------------------------------------------------------
// Kernel 4: logits = hid @ W3 + b3 (f32 out). N=16 -> one WMMA tile per wave.
// Grid: 64 blocks x 128 rows, wave = M-tile.
// ---------------------------------------------------------------------------
__global__ __launch_bounds__(256) void k_actor2(const _Float16* __restrict__ hidf,
                                                const _Float16* __restrict__ w3t,
                                                const float* __restrict__ b3,
                                                float* __restrict__ out) {
  const int tid = threadIdx.x, lane = tid & 31, wave = tid >> 5;
  const int hi = lane >> 4, r16 = lane & 15;
  const int mBlock = blockIdx.x * 128;
  const int row = mBlock + wave * 16 + r16;

  v8f acc = vzero8();
#pragma unroll
  for (int s = 0; s < HID / 32; ++s) {
    const int k0 = s * 32;
    v16h a = afrag_f16(hidf + (size_t)row * HID + k0 + hi * 8);
    v16h b = bfrag_kmaj(w3t + r16 * HID + k0 + hi * 8);
    acc = wmma32(a, b, acc);
  }
  const float bv = b3[r16];
#pragma unroll
  for (int rr = 0; rr < 8; ++rr) {
    const int gr = mBlock + wave * 16 + hi * 8 + rr;
    out[(size_t)gr * ACT_DIM + r16] = acc[rr] + bv;
  }
}

// ---------------------------------------------------------------------------
extern "C" void kernel_launch(void* const* d_in, const int* in_sizes, int n_in,
                              void* d_out, int out_size, void* d_ws,
                              size_t ws_size, hipStream_t stream) {
  (void)in_sizes; (void)n_in; (void)out_size; (void)ws_size;

  const float* obs = (const float*)d_in[0];
  const int* adj = (const int*)d_in[1];
  const float* W1 = (const float*)d_in[2];
  const float* b1 = (const float*)d_in[3];
  const float* W2 = (const float*)d_in[4];
  const float* b2 = (const float*)d_in[5];
  const float* W3 = (const float*)d_in[6];
  const float* b3 = (const float*)d_in[7];
  float* out = (float*)d_out;

  // workspace carve-up (all 256B-aligned): 4 x 2MB f16 activations + weights
  char* w = (char*)d_ws;
  _Float16* hrow = (_Float16*)(w);                               // 2 MB
  _Float16* hT   = (_Float16*)(w + (size_t)(2u << 20));          // 2 MB
  _Float16* msgf = (_Float16*)(w + (size_t)(4u << 20));          // 2 MB
  _Float16* hidf = (_Float16*)(w + (size_t)(6u << 20));          // 2 MB
  _Float16* w1t  = (_Float16*)(w + (size_t)(8u << 20));          // 16 KB
  _Float16* w2t  = (_Float16*)(w + (size_t)(8u << 20) + 32 * 1024);   // 64 KB
  _Float16* w3t  = (_Float16*)(w + (size_t)(8u << 20) + 128 * 1024);  // 4 KB

  k_prep<<<64, 256, 0, stream>>>(W1, W2, W3, w1t, w2t, w3t);
  k_encoder<<<N_AGENTS / 64, 256, 0, stream>>>(obs, w1t, b1, hrow, hT);
  k_message<<<N_AGENTS / 64, 256, 0, stream>>>(adj, hT, msgf);
  k_actor1<<<N_AGENTS / 64, 256, 0, stream>>>(hrow, msgf, w2t, b2, hidf);
  k_actor2<<<N_AGENTS / 128, 256, 0, stream>>>(hidf, w3t, b3, out);
}
